// DynamicToepliztMultiheadV2_9646496546861
// MI455X (gfx1250) — matrix operand order repaired and checked by reading, not compiled, not run on hardware
//
#include <hip/hip_runtime.h>

typedef __attribute__((ext_vector_type(16))) _Float16 v16h;
typedef __attribute__((ext_vector_type(8)))  _Float16 v8h;
typedef __attribute__((ext_vector_type(8)))  float    v8f;
typedef __attribute__((ext_vector_type(4)))  float    v4f;
typedef __attribute__((ext_vector_type(4)))  unsigned int v4u;

#define B_    8
#define H_    8
#define N_    4096
#define E_    64
#define PD_   16
#define TWO_N (2 * N_)
#define LN_EPSF 1e-5f

// ---------------------------------------------------------------------------
// Kernel A: build reversed Toeplitz generator table.
//   gtab[h * TWO_N + u] = dpb(N_ - u)[h]   (f16), u in [0, 2n)
// Used entries are u in [1, 2n-1]  <->  d = i-j in [-(n-1), n-1].
// ---------------------------------------------------------------------------
__device__ __forceinline__ void ln_relu16(float* v, const float* g, const float* be) {
    float mu = 0.f;
#pragma unroll
    for (int p = 0; p < 16; ++p) mu += v[p];
    mu *= (1.f / 16.f);
    float var = 0.f;
#pragma unroll
    for (int p = 0; p < 16; ++p) { float d = v[p] - mu; var += d * d; }
    var *= (1.f / 16.f);
    float rs = rsqrtf(var + LN_EPSF);
#pragma unroll
    for (int p = 0; p < 16; ++p) {
        float t = (v[p] - mu) * rs * g[p] + be[p];
        v[p] = t > 0.f ? t : 0.f;
    }
}

__global__ __launch_bounds__(256) void dpb_table_kernel(
    const float* __restrict__ W0, const float* __restrict__ b0,
    const float* __restrict__ g1, const float* __restrict__ be1,
    const float* __restrict__ W1, const float* __restrict__ b1,
    const float* __restrict__ g2, const float* __restrict__ be2,
    const float* __restrict__ W2, const float* __restrict__ b2,
    const float* __restrict__ g3, const float* __restrict__ be3,
    const float* __restrict__ W3, const float* __restrict__ b3,
    _Float16* __restrict__ gtab)
{
    int u = blockIdx.x * blockDim.x + threadIdx.x;
    if (u >= TWO_N) return;
    float t = (float)(N_ - u);

    float ha[16], hb[16];
#pragma unroll
    for (int p = 0; p < 16; ++p) ha[p] = t * W0[p] + b0[p];
    ln_relu16(ha, g1, be1);

#pragma unroll
    for (int q = 0; q < 16; ++q) hb[q] = b1[q];
    for (int p = 0; p < 16; ++p) {
        float v = ha[p];
#pragma unroll
        for (int q = 0; q < 16; ++q) hb[q] += v * W1[p * 16 + q];
    }
    ln_relu16(hb, g2, be2);

#pragma unroll
    for (int q = 0; q < 16; ++q) ha[q] = b2[q];
    for (int p = 0; p < 16; ++p) {
        float v = hb[p];
#pragma unroll
        for (int q = 0; q < 16; ++q) ha[q] += v * W2[p * 16 + q];
    }
    ln_relu16(ha, g3, be3);

    float o[H_];
#pragma unroll
    for (int hh = 0; hh < H_; ++hh) o[hh] = b3[hh];
    for (int p = 0; p < 16; ++p) {
        float v = ha[p];
#pragma unroll
        for (int hh = 0; hh < H_; ++hh) o[hh] += v * W3[p * H_ + hh];
    }
#pragma unroll
    for (int hh = 0; hh < H_; ++hh)
        gtab[(size_t)hh * TWO_N + u] = (_Float16)o[hh];
}

// ---------------------------------------------------------------------------
// Kernel B: out[b,h] (4096 x 64) = Toeplitz(f_h) @ x[b,h], WMMA f16->f32.
// Block: 256 threads (8 waves, wave32). Output tile: 128 rows x 64 cols.
// Wave w -> (mr = w>>1, mc = w&1) computes a 32x32 macro-tile: 2x2 WMMA
// C tiles held in registers across the whole K loop (4 x v8f accumulators).
// Per k-step: 2 A-fragment gathers + 2 B-fragment loads feed 4 WMMAs.
// ---------------------------------------------------------------------------
#define XS_STRIDE 72   // halves; 144 B rows: 16B-aligned, conflict-free b128 reads
#define ROWS_PB  128   // output rows per block
#define NCHUNK   (N_ / 64)

__global__ __launch_bounds__(256) void toeplitz_wmma_kernel(
    const float* __restrict__ x, const _Float16* __restrict__ gtab_g,
    float* __restrict__ out)
{
    __shared__ _Float16 gt[TWO_N];                                  // 16 KB
    __shared__ __align__(16) _Float16 xs[2][E_][XS_STRIDE];         // 18 KB

    const int itile = blockIdx.x;            // row tile (L2 reuse: fastest dim)
    const int h = blockIdx.y, b = blockIdx.z;
    const int tid = threadIdx.x;
    const int i0 = itile * ROWS_PB;

    const float* xp = x + (size_t)(b * H_ + h) * N_ * E_;
    float*       op = out + (size_t)(b * H_ + h) * N_ * E_;

    // Stage whole reversed head table into LDS (8192 halves = 1024 x 16B).
    {
        const v4u* src = (const v4u*)(gtab_g + (size_t)h * TWO_N);
        v4u* dst = (v4u*)gt;
#pragma unroll
        for (int k = 0; k < 4; ++k) dst[tid + 256 * k] = src[tid + 256 * k];
    }

    const int lane = tid & 31, w = tid >> 5;
    const int lm = lane & 15, lh = lane >> 4;
    const int mr = w >> 1, mc = w & 1;
    const int Mrow0 = i0 + mr * 32 + lm;     // A-frag row, upper 16x16 row tile
    const int Mrow1 = Mrow0 + 16;            // lower row tile
    const int eb0 = mc * 32, eb1 = mc * 32 + 16;

    v8f c00 = {}, c01 = {}, c10 = {}, c11 = {};

    // staging map: 4 consecutive e, 4 consecutive j rows per thread
    const int se = (tid & 15) * 4;
    const int sj = (tid >> 4) * 4;
    float xr[16];

    auto stage_load = [&](int jc) {
        const float* p = xp + (size_t)(jc * 64 + sj) * E_ + se;
#pragma unroll
        for (int s = 0; s < 4; ++s) {
            v4f v = *(const v4f*)(p + (size_t)s * E_);
#pragma unroll
            for (int q = 0; q < 4; ++q) xr[s * 4 + q] = v[q];
        }
    };
    auto stage_store = [&](int jc) {
        const int buf = jc & 1;
#pragma unroll
        for (int q = 0; q < 4; ++q)
#pragma unroll
            for (int s = 0; s < 4; ++s)
                xs[buf][se + q][sj + s] = (_Float16)xr[s * 4 + q];
    };
    auto gather_a = [&](int ub) {
        // T[i][j] = gt[(j - i) + N_]; 16-bit 16x32 A layout matching B:
        // per lane, 16 consecutive K (K = lh*16 + p).
        v16h a;
#pragma unroll
        for (int p = 0; p < 16; ++p) {
            const int K = (lh << 4) + p;
            a[p] = gt[ub + K];
        }
        return a;
    };
    auto compute = [&](int jc) {
        const int buf = jc & 1;
#pragma unroll
        for (int ks = 0; ks < 2; ++ks) {
            const int kbase = jc * 64 + ks * 32;
            const v16h a0 = gather_a(kbase + N_ - Mrow0);
            const v16h a1 = gather_a(kbase + N_ - Mrow1);
            // B fragments: 16 consecutive K-halves per lane (K = lh*16 + p)
            const int koff = ks * 32 + (lh << 4);
            const v8h* r0 = (const v8h*)&xs[buf][eb0 + lm][koff];
            const v8h* r1 = (const v8h*)&xs[buf][eb1 + lm][koff];
            const v8h b0a = r0[0], b0b = r0[1];
            const v8h b1a = r1[0], b1b = r1[1];
            v16h bf0, bf1;
#pragma unroll
            for (int p = 0; p < 8; ++p) {
                bf0[p] = b0a[p]; bf0[p + 8] = b0b[p];
                bf1[p] = b1a[p]; bf1[p + 8] = b1b[p];
            }
            c00 = __builtin_amdgcn_wmma_f32_16x16x32_f16(
                      false, a0, false, bf0, (short)0, c00, false, false);
            c01 = __builtin_amdgcn_wmma_f32_16x16x32_f16(
                      false, a0, false, bf1, (short)0, c01, false, false);
            c10 = __builtin_amdgcn_wmma_f32_16x16x32_f16(
                      false, a1, false, bf0, (short)0, c10, false, false);
            c11 = __builtin_amdgcn_wmma_f32_16x16x32_f16(
                      false, a1, false, bf1, (short)0, c11, false, false);
        }
    };

    stage_load(0);
    stage_store(0);
    __syncthreads();   // table + chunk 0 visible

    for (int jc = 0; jc < NCHUNK; ++jc) {
        if (jc + 1 < NCHUNK) {
            stage_load(jc + 1);                                   // global -> regs
            const int pj = (jc + 2 < NCHUNK) ? jc + 2 : jc + 1;
            __builtin_prefetch(xp + (size_t)(pj * 64 + sj) * E_ + se, 0, 0);
        }
        compute(jc);                                              // reads buf jc&1
        if (jc + 1 < NCHUNK) stage_store(jc + 1);                 // writes other buf
        __syncthreads();
    }

    // C/D layout: VGPR r -> row (lh?8:0)+r, lane lm -> col
#pragma unroll
    for (int r = 0; r < 8; ++r) {
        const int ia = i0 + mr * 32 + (lh ? 8 : 0) + r;
        const int ib = ia + 16;
        op[(size_t)ia * E_ + eb0 + lm] = c00[r];
        op[(size_t)ia * E_ + eb1 + lm] = c01[r];
        op[(size_t)ib * E_ + eb0 + lm] = c10[r];
        op[(size_t)ib * E_ + eb1 + lm] = c11[r];
    }
}

// ---------------------------------------------------------------------------
extern "C" void kernel_launch(void* const* d_in, const int* in_sizes, int n_in,
                              void* d_out, int out_size, void* d_ws, size_t ws_size,
                              hipStream_t stream) {
    const float* x   = (const float*)d_in[0];
    const float* W0  = (const float*)d_in[1];
    const float* b0  = (const float*)d_in[2];
    const float* g1  = (const float*)d_in[3];
    const float* be1 = (const float*)d_in[4];
    const float* W1  = (const float*)d_in[5];
    const float* b1  = (const float*)d_in[6];
    const float* g2  = (const float*)d_in[7];
    const float* be2 = (const float*)d_in[8];
    const float* W2  = (const float*)d_in[9];
    const float* b2  = (const float*)d_in[10];
    const float* g3  = (const float*)d_in[11];
    const float* be3 = (const float*)d_in[12];
    const float* W3  = (const float*)d_in[13];
    const float* b3  = (const float*)d_in[14];

    _Float16* gtab = (_Float16*)d_ws;   // H_ * TWO_N halves = 128 KB

    dpb_table_kernel<<<TWO_N / 256, 256, 0, stream>>>(
        W0, b0, g1, be1, W1, b1, g2, be2, W2, b2, g3, be3, W3, b3, gtab);

    dim3 grid(N_ / ROWS_PB, H_, B_);
    toeplitz_wmma_kernel<<<grid, 256, 0, stream>>>(x, gtab, (float*)d_out);
}
